// Model_76304388980993
// MI455X (gfx1250) — matrix operand order repaired and checked by reading, not compiled, not run on hardware
//
#include <hip/hip_runtime.h>
#include <hip/hip_bf16.h>
#include <math.h>

typedef __attribute__((ext_vector_type(16))) _Float16 v16h;
typedef __attribute__((ext_vector_type(8)))  _Float16 v8h;
typedef __attribute__((ext_vector_type(8)))  float    v8f;

union AFrag { v16h v; v8h h[2]; };

// CDNA5 hardware V_TANH_F32 transcendental (verified in round-2/3 disasm).
__device__ __forceinline__ float fast_tanh(float x) {
#if __has_builtin(__builtin_amdgcn_tanhf)
    return __builtin_amdgcn_tanhf(x);
#elif __has_builtin(__builtin_amdgcn_tanh_f32)
    return __builtin_amdgcn_tanh_f32(x);
#else
    return tanhf(x);
#endif
}

// ---------------------------------------------------------------------------
// WMMA message-passing kernel:  agg[dst] += tanh([h_src[src] | ef] @ W + b)
// A = 16 edges x 96 (K: 64 node feats + 8 edge feats + 24 zero pad)
// B = W[72,64] zero-padded to 96x64, held in registers in WMMA layout
// C = 16 x 64 f32, 4 N-tiles of v8f accumulators
// SMALL=true  -> destinations are devices (<=48): block-private LDS accum
// SMALL=false -> destinations are ops (20000):  global f32 atomics
// ---------------------------------------------------------------------------
template<bool SMALL>
__global__ void __launch_bounds__(128) k_mp(
    const _Float16* __restrict__ hsrc,   // [nsrc][64] f16 mirror of node state
    const _Float16* __restrict__ ef,     // [E][8] transformed edge feats, f16
    const int* __restrict__ src,
    const int* __restrict__ dst,
    int E,
    const float* __restrict__ W,         // [72][64] row-major (k, n)
    const float* __restrict__ bias,      // [64]
    float* __restrict__ agg,             // [ndst][64] f32
    int aggN)                            // ndst*64 (flush size for SMALL)
{
    __shared__ __align__(32) _Float16 a_lds[4][16][96];   // per-wave A tile
    __shared__ int   dst_lds[4][16];
    __shared__ float devacc[3072];                        // 48*64 partials

    const int tid  = threadIdx.x;
    const int lane = tid & 31;
    const int w    = tid >> 5;

    if (SMALL) {
        for (int i = tid; i < 3072; i += 128) devacc[i] = 0.f;
        __syncthreads();
    }

    // --- preload B fragments (weights) in WMMA 32x16 f16 layout ------------
    // lane l: column n = l&15 ; half j holds K = (l>>4)*8 + (j&7) + (j>>3)*16
    const int bn  = lane & 15;
    const int bkb = (lane >> 4) << 3;
    v16h bfrag[3][4];
#pragma unroll
    for (int kc = 0; kc < 3; ++kc)
#pragma unroll
        for (int nt = 0; nt < 4; ++nt) {
            v16h b;
#pragma unroll
            for (int j = 0; j < 16; ++j) {
                int k = kc * 32 + bkb + (j & 7) + ((j >> 3) << 4);
                float wv = (k < 72) ? W[k * 64 + nt * 16 + bn] : 0.f;
                b[j] = (_Float16)wv;
            }
            bfrag[kc][nt] = b;
        }

    float biav[4];
#pragma unroll
    for (int nt = 0; nt < 4; ++nt) biav[nt] = bias[nt * 16 + bn];

    const int arow = lane & 15;         // A frag: row = lane&15
    const int akb  = (lane >> 4) << 3;  //         K base 0 or 8 per half-wave
    const int srow = lane >> 1;         // staging: 2 lanes per edge row
    const int ssub = lane & 1;

    const int ntiles = (E + 15) >> 4;
    const int wave   = blockIdx.x * 4 + w;
    const int nwaves = gridDim.x * 4;

    for (int t = wave; t < ntiles; t += nwaves) {
        // ---- stage A tile into LDS: h_src(64) | ef(8) | zeros(24) --------
        {
            int  e  = t * 16 + srow;
            bool ok = e < E;
            int  s  = ok ? src[e] : 0;
            uint4* ap = (uint4*)&a_lds[w][srow][0];            // 12 slots
            const uint4* hp = (const uint4*)(hsrc + (size_t)s * 64);
#pragma unroll
            for (int i = 0; i < 4; ++i) ap[ssub * 4 + i] = hp[ssub * 4 + i];
            uint4 z = make_uint4(0u, 0u, 0u, 0u);
            if (ssub == 0) {
                uint4 ev = z;
                if (ok) ev = *(const uint4*)(ef + (size_t)e * 8);
                ap[8] = ev;
                dst_lds[w][srow] = ok ? dst[e] : 0;
            } else {
                ap[9] = z; ap[10] = z; ap[11] = z;
            }
        }
        // cross-lane LDS write -> read ordering (same wave, LDS in-order)
        asm volatile("s_wait_dscnt 0" ::: "memory");

        // ---- WMMA: [16x96] x [96x64] -> [16x64] f32 ----------------------
        v8f z8 = {0.f,0.f,0.f,0.f,0.f,0.f,0.f,0.f};
        v8f acc[4] = {z8, z8, z8, z8};
#pragma unroll
        for (int kc = 0; kc < 3; ++kc) {
            AFrag a;
            a.h[0] = *(const v8h*)&a_lds[w][arow][kc * 32 + akb];
            a.h[1] = *(const v8h*)&a_lds[w][arow][kc * 32 + 16 + akb];
#pragma unroll
            for (int nt = 0; nt < 4; ++nt)
                acc[nt] = __builtin_amdgcn_wmma_f32_16x16x32_f16(
                    false, a.v, false, bfrag[kc][nt],
                    (short)0, acc[nt], false, false);
        }

        // ---- bias + hardware tanh + scatter-add --------------------------
        // C layout: lane l, vgpr r -> (M = r + 8*(l>>4), N = l&15)
        // r-outer / nt-inner with unsigned 32-bit element indices: lets the
        // backend use SADDR + 32-bit voffset addressing for the atomics and
        // fold the nt*16 stride into the instruction immediate offset.
        const int mbase = (lane >> 4) << 3;
        if (t * 16 + 16 <= E) {
            // fast path: full tile, no per-element guards
#pragma unroll
            for (int r = 0; r < 8; ++r) {
                unsigned d   = (unsigned)dst_lds[w][mbase + r];
                unsigned idx = d * 64u + (unsigned)bn;
#pragma unroll
                for (int nt = 0; nt < 4; ++nt) {
                    float val = fast_tanh(acc[nt][r] + biav[nt]);
                    if (SMALL) atomicAdd(&devacc[idx + nt * 16], val);
                    else       atomicAdd(&agg[idx + nt * 16], val);
                }
            }
        } else {
            // tail tile: guard each edge row
#pragma unroll
            for (int r = 0; r < 8; ++r) {
                int M = mbase + r;
                if (t * 16 + M < E) {
                    unsigned d   = (unsigned)dst_lds[w][M];
                    unsigned idx = d * 64u + (unsigned)bn;
#pragma unroll
                    for (int nt = 0; nt < 4; ++nt) {
                        float val = fast_tanh(acc[nt][r] + biav[nt]);
                        if (SMALL) atomicAdd(&devacc[idx + nt * 16], val);
                        else       atomicAdd(&agg[idx + nt * 16], val);
                    }
                }
            }
        }
    }

    if (SMALL) {
        __syncthreads();
        for (unsigned i = tid; i < (unsigned)aggN; i += 128u)
            atomicAdd(&agg[i], devacc[i]);
    }
}

// ---------------------------------------------------------------------------
// small helper kernels (each < 2% of total FLOPs, L2-resident)
// ---------------------------------------------------------------------------
__global__ void k_zero(float* p, int n) {
    int i = blockIdx.x * blockDim.x + threadIdx.x;
    if (i < n) p[i] = 0.f;
}

// ef = tanh(raw[E,F] @ W[F,8] + b)  stored f16
__global__ void k_edge_tr(const float* __restrict__ raw, int E, int F,
                          const float* __restrict__ W, const float* __restrict__ b,
                          _Float16* __restrict__ ef) {
    int e = blockIdx.x * blockDim.x + threadIdx.x;
    if (e >= E) return;
    const float* r = raw + (size_t)e * F;
    float o[8];
#pragma unroll
    for (int j = 0; j < 8; ++j) o[j] = b[j];
    for (int k = 0; k < F; ++k) {
        float x = r[k];
#pragma unroll
        for (int j = 0; j < 8; ++j) o[j] += x * W[k * 8 + j];
    }
    _Float16* out = ef + (size_t)e * 8;
#pragma unroll
    for (int j = 0; j < 8; ++j) out[j] = (_Float16)fast_tanh(o[j]);
}

// op_h = tanh([op_feats | emb[op_types]] @ W[F+8,64] + b); one block per op
__global__ void k_op_trans(const float* __restrict__ x, const float* __restrict__ emb,
                           const int* __restrict__ types, int F,
                           const float* __restrict__ W, const float* __restrict__ b,
                           float* __restrict__ h, _Float16* __restrict__ h16) {
    int i = blockIdx.x, n = threadIdx.x;
    float acc = b[n];
    const float* xr = x + (size_t)i * F;
    for (int k = 0; k < F; ++k) acc += xr[k] * W[k * 64 + n];
    const float* er = emb + (size_t)types[i] * 8;
#pragma unroll
    for (int j = 0; j < 8; ++j) acc += er[j] * W[(F + j) * 64 + n];
    float v = fast_tanh(acc);
    h  [(size_t)i * 64 + n] = v;
    h16[(size_t)i * 64 + n] = (_Float16)v;
}

__global__ void k_dev_trans(const float* __restrict__ x, int F,
                            const float* __restrict__ W, const float* __restrict__ b,
                            float* __restrict__ h, _Float16* __restrict__ h16) {
    int i = blockIdx.x, n = threadIdx.x;
    float acc = b[n];
    const float* xr = x + (size_t)i * F;
    for (int k = 0; k < F; ++k) acc += xr[k] * W[k * 64 + n];
    float v = fast_tanh(acc);
    h  [(size_t)i * 64 + n] = v;
    h16[(size_t)i * 64 + n] = (_Float16)v;
}

// h += agg @ W[64,64] + b ; refresh f16 mirror
__global__ void k_update(float* __restrict__ h, _Float16* __restrict__ h16,
                         const float* __restrict__ agg,
                         const float* __restrict__ W, const float* __restrict__ b) {
    int i = blockIdx.x, n = threadIdx.x;
    float acc = b[n];
    const float* ar = agg + (size_t)i * 64;
    for (int k = 0; k < 64; ++k) acc += ar[k] * W[k * 64 + n];
    float v = h[(size_t)i * 64 + n] + acc;
    h  [(size_t)i * 64 + n] = v;
    h16[(size_t)i * 64 + n] = (_Float16)v;
}

// out[i,d] = op_h[i,:] . dev_h[d,:]
__global__ void k_out(const float* __restrict__ oph, const float* __restrict__ dvh,
                      float* __restrict__ out, int ndev) {
    int i = blockIdx.x, d = threadIdx.x;
    if (d >= ndev) return;
    const float* a  = oph + (size_t)i * 64;
    const float* bb = dvh + (size_t)d * 64;
    float acc = 0.f;
    for (int k = 0; k < 64; ++k) acc += a[k] * bb[k];
    out[(size_t)i * ndev + d] = acc;
}

// ---------------------------------------------------------------------------
extern "C" void kernel_launch(void* const* d_in, const int* in_sizes, int n_in,
                              void* d_out, int out_size, void* d_ws, size_t ws_size,
                              hipStream_t stream) {
    (void)n_in; (void)out_size; (void)ws_size;
    // ---- inputs in setup_inputs() dict order, params pytree in insertion order
    const float* op_feats    = (const float*)d_in[0];
    const float* dev_feats   = (const float*)d_in[1];
    const float* tensor_f    = (const float*)d_in[2];
    const float* link_f      = (const float*)d_in[3];
    const float* place_f     = (const float*)d_in[4];
    const float* emb         = (const float*)d_in[5];
    const float* opW  = (const float*)d_in[6];
    const float* opB  = (const float*)d_in[7];
    const float* dvW  = (const float*)d_in[8];
    const float* dvB  = (const float*)d_in[9];
    const float* etW[5]; const float* etB[5];          // link,prev,succ,place,serve
    for (int i = 0; i < 5; ++i) { etW[i] = (const float*)d_in[10 + 2*i];
                                  etB[i] = (const float*)d_in[11 + 2*i]; }
    struct Layer { const float *eW[5], *eB[5], *ofW, *ofB, *dfW, *dfB; } L[3];
    int p = 20;
    for (int l = 0; l < 3; ++l) {
        for (int i = 0; i < 5; ++i) { L[l].eW[i] = (const float*)d_in[p++];
                                      L[l].eB[i] = (const float*)d_in[p++]; }
        L[l].ofW = (const float*)d_in[p++]; L[l].ofB = (const float*)d_in[p++];
        L[l].dfW = (const float*)d_in[p++]; L[l].dfB = (const float*)d_in[p++];
    }
    const int* op_types  = (const int*)d_in[62];
    const int* prev_src  = (const int*)d_in[63]; const int* prev_dst  = (const int*)d_in[64];
    const int* succ_src  = (const int*)d_in[65]; const int* succ_dst  = (const int*)d_in[66];
    const int* place_src = (const int*)d_in[67]; const int* place_dst = (const int*)d_in[68];
    const int* serve_src = (const int*)d_in[69]; const int* serve_dst = (const int*)d_in[70];
    const int* link_src  = (const int*)d_in[71]; const int* link_dst  = (const int*)d_in[72];

    // ---- sizes derived from in_sizes (robust to N changes)
    const int N_OP  = in_sizes[62];
    const int F_DEV = in_sizes[8] / 64;
    const int N_DEV = in_sizes[1] / F_DEV;
    const int F_OP  = in_sizes[6] / 64 - 8;
    const int E_T   = in_sizes[63];
    const int E_P   = in_sizes[67];
    const int E_L   = in_sizes[71];
    const int F_T   = in_sizes[2] / E_T;
    const int F_L   = in_sizes[3] / E_L;
    const int F_P   = in_sizes[4] / E_P;

    // ---- workspace carve-up (256B aligned)
    char* ws = (char*)d_ws; size_t off = 0;
    auto carve = [&](size_t bytes) -> void* {
        void* r = ws + off; off = (off + bytes + 255) & ~(size_t)255; return r;
    };
    float*     op_h   = (float*)carve((size_t)N_OP * 64 * 4);
    float*     op_agg = (float*)carve((size_t)N_OP * 64 * 4);
    _Float16*  op_h16 = (_Float16*)carve((size_t)N_OP * 64 * 2);
    float*     dv_h   = (float*)carve((size_t)N_DEV * 64 * 4);
    float*     dv_agg = (float*)carve((size_t)N_DEV * 64 * 4);
    _Float16*  dv_h16 = (_Float16*)carve((size_t)N_DEV * 64 * 2);
    _Float16*  ef_link  = (_Float16*)carve((size_t)E_L * 8 * 2);
    _Float16*  ef_prev  = (_Float16*)carve((size_t)E_T * 8 * 2);
    _Float16*  ef_succ  = (_Float16*)carve((size_t)E_T * 8 * 2);
    _Float16*  ef_place = (_Float16*)carve((size_t)E_P * 8 * 2);
    _Float16*  ef_serve = (_Float16*)carve((size_t)E_P * 8 * 2);

    auto mp_grid = [](int E) {
        int nt = (E + 15) / 16; int b = (nt + 3) / 4;
        if (b > 4096) b = 4096; if (b < 1) b = 1; return b;
    };
    const bool small_dev = (N_DEV * 64 <= 3072);

    // ---- input transforms (once; edge transforms are layer-invariant)
    k_op_trans <<<N_OP,  64, 0, stream>>>(op_feats, emb, op_types, F_OP, opW, opB, op_h, op_h16);
    k_dev_trans<<<N_DEV, 64, 0, stream>>>(dev_feats, F_DEV, dvW, dvB, dv_h, dv_h16);
    k_edge_tr<<<(E_L + 255) / 256, 256, 0, stream>>>(link_f,   E_L, F_L, etW[0], etB[0], ef_link);
    k_edge_tr<<<(E_T + 255) / 256, 256, 0, stream>>>(tensor_f, E_T, F_T, etW[1], etB[1], ef_prev);
    k_edge_tr<<<(E_T + 255) / 256, 256, 0, stream>>>(tensor_f, E_T, F_T, etW[2], etB[2], ef_succ);
    k_edge_tr<<<(E_P + 255) / 256, 256, 0, stream>>>(place_f,  E_P, F_P, etW[3], etB[3], ef_place);
    k_edge_tr<<<(E_P + 255) / 256, 256, 0, stream>>>(place_f,  E_P, F_P, etW[4], etB[4], ef_serve);

    // ---- 3 graph-conv layers
    for (int l = 0; l < 3; ++l) {
        k_zero<<<(N_OP  * 64 + 255) / 256, 256, 0, stream>>>(op_agg, N_OP  * 64);
        k_zero<<<(N_DEV * 64 + 255) / 256, 256, 0, stream>>>(dv_agg, N_DEV * 64);

        // op-side aggregation: prev + succ + serve
        k_mp<false><<<mp_grid(E_T), 128, 0, stream>>>(op_h16, ef_prev, prev_src, prev_dst,
            E_T, L[l].eW[1], L[l].eB[1], op_agg, N_OP * 64);
        k_mp<false><<<mp_grid(E_T), 128, 0, stream>>>(op_h16, ef_succ, succ_src, succ_dst,
            E_T, L[l].eW[2], L[l].eB[2], op_agg, N_OP * 64);
        k_mp<false><<<mp_grid(E_P), 128, 0, stream>>>(dv_h16, ef_serve, serve_src, serve_dst,
            E_P, L[l].eW[4], L[l].eB[4], op_agg, N_OP * 64);

        // device-side aggregation: link + place (48 hot rows -> LDS accum path)
        if (small_dev) {
            k_mp<true><<<mp_grid(E_L), 128, 0, stream>>>(dv_h16, ef_link, link_src, link_dst,
                E_L, L[l].eW[0], L[l].eB[0], dv_agg, N_DEV * 64);
            k_mp<true><<<mp_grid(E_P), 128, 0, stream>>>(op_h16, ef_place, place_src, place_dst,
                E_P, L[l].eW[3], L[l].eB[3], dv_agg, N_DEV * 64);
        } else {
            k_mp<false><<<mp_grid(E_L), 128, 0, stream>>>(dv_h16, ef_link, link_src, link_dst,
                E_L, L[l].eW[0], L[l].eB[0], dv_agg, N_DEV * 64);
            k_mp<false><<<mp_grid(E_P), 128, 0, stream>>>(op_h16, ef_place, place_src, place_dst,
                E_P, L[l].eW[3], L[l].eB[3], dv_agg, N_DEV * 64);
        }

        k_update<<<N_OP,  64, 0, stream>>>(op_h, op_h16, op_agg, L[l].ofW, L[l].ofB);
        k_update<<<N_DEV, 64, 0, stream>>>(dv_h, dv_h16, dv_agg, L[l].dfW, L[l].dfB);
    }

    // ---- output: op_h @ dev_h.T
    int blk = ((N_DEV + 31) / 32) * 32; if (blk < 32) blk = 32;
    k_out<<<N_OP, blk, 0, stream>>>(op_h, dv_h, (float*)d_out, N_DEV);
}
